// GraphRecurrent_54107997995648
// MI455X (gfx1250) — matrix operand order repaired and checked by reading, not compiled, not run on hardware
//
#include <hip/hip_runtime.h>

typedef __attribute__((ext_vector_type(16))) _Float16 v16h;
typedef __attribute__((ext_vector_type(8)))  _Float16 h8;
typedef __attribute__((ext_vector_type(8)))  float    v8f;
typedef __attribute__((ext_vector_type(4)))  float    f4;

#define NN    96
#define DD    64
#define BSZ   16
#define TT    13
#define ROWS  1536
#define FRAG  512       // halves per fragment: 32 lanes x 16
#define WSTR  4096      // halves per packed 64x64 weight: 8 fragments

__device__ __forceinline__ v8f zero8() {
    v8f z;
#pragma unroll
    for (int i = 0; i < 8; ++i) z[i] = 0.0f;
    return z;
}

__device__ __forceinline__ v8f wmma16(v16h a, v16h b, v8f c) {
    return __builtin_amdgcn_wmma_f32_16x16x32_f16(
        false, a, false, b, (short)0, c, false, false);
}

// Packed B fragment: one aligned 32-byte load per lane
__device__ __forceinline__ v16h load_bp(const _Float16* __restrict__ P,
                                        int nt, int kh, int lane) {
    return *(const v16h*)(P + (nt * 2 + kh) * FRAG + lane * 16);
}

// A fragment from row-major f32 16x64 tile (row stride 64): 4x b128 + cvt
// A packing (ISA 7.12.2): lane m=lane%16; e<8 -> k=e+8*hs, e>=8 -> k=e+8+8*hs (+32*kh)
__device__ __forceinline__ v16h load_af32(const float* __restrict__ M, int kh, int lane) {
    int m = lane & 15, hs = lane >> 4;
    const float* p = M + m * DD + kh * 32 + 8 * hs;
    f4 a0 = *(const f4*)(p);
    f4 a1 = *(const f4*)(p + 4);
    f4 a2 = *(const f4*)(p + 16);
    f4 a3 = *(const f4*)(p + 20);
    v16h f;
#pragma unroll
    for (int e = 0; e < 4; ++e) {
        f[e]      = (_Float16)a0[e];
        f[e + 4]  = (_Float16)a1[e];
        f[e + 8]  = (_Float16)a2[e];
        f[e + 12] = (_Float16)a3[e];
    }
    return f;
}

// A fragment from row-major f16 16x64 LDS tile: 2x 16-byte ds loads
__device__ __forceinline__ v16h load_alds(const _Float16* __restrict__ L, int kh, int lane) {
    int m = lane & 15, hs = lane >> 4;
    const _Float16* p = L + m * DD + kh * 32 + 8 * hs;
    h8 x0 = *(const h8*)(p);
    h8 x1 = *(const h8*)(p + 16);
    v16h f;
#pragma unroll
    for (int e = 0; e < 8; ++e) { f[e] = x0[e]; f[e + 8] = x1[e]; }
    return f;
}

// ---------------------------------------------------------------------------
// Pre-pack all weight matrices into f16 WMMA B fragments.
// B layout (ISA): lane n=lane%16 holds column n; e<8 -> k=e+8*hs, e>=8 -> k=e+8+8*hs (+32*kh).
// grid 9*8 = 72 blocks of 32
__global__ void k_pack(const float* __restrict__ Wm1, const float* __restrict__ Wm2,
                       const float* __restrict__ Whr, const float* __restrict__ Whi,
                       const float* __restrict__ Whh, const float* __restrict__ Wo1,
                       const float* __restrict__ Wo2, const float* __restrict__ Wo3,
                       _Float16* __restrict__ dst) {
    int w = blockIdx.x >> 3;
    int f = blockIdx.x & 7;           // nt*2 + kh
    int nt = f >> 1, kh = f & 1;
    int lane = threadIdx.x;
    int n = lane & 15, hs = lane >> 4;
    const float* W;
    switch (w) {
        case 0: W = Wm1;            break;   // sender rows 0..63
        case 1: W = Wm1 + 64 * DD;  break;   // receiver rows 64..127
        case 2: W = Wm2;            break;
        case 3: W = Whr;            break;
        case 4: W = Whi;            break;
        case 5: W = Whh;            break;
        case 6: W = Wo1;            break;
        case 7: W = Wo2;            break;
        default: W = Wo3;           break;
    }
    v16h v;
#pragma unroll
    for (int e = 0; e < 16; ++e) {
        int k = (e < 8 ? e + 8 * hs : e + 8 + 8 * hs) + 32 * kh;
        v[e] = (_Float16)W[k * DD + nt * 16 + n];
    }
    *(v16h*)(dst + w * WSTR + f * FRAG + lane * 16) = v;
}

// ---------------------------------------------------------------------------
// Hs = hidden @ Wm1[0:64], Hr = hidden @ Wm1[64:128]
__global__ void k_hsr(const float* __restrict__ hidden,
                      const _Float16* __restrict__ Pm1s, const _Float16* __restrict__ Pm1r,
                      float* __restrict__ Hs, float* __restrict__ Hr) {
    int tile = blockIdx.x;
    int lane = threadIdx.x;
    int n = lane & 15, hs = lane >> 4;

    const float* rows = hidden + tile * 16 * DD;
    v16h A0 = load_af32(rows, 0, lane);
    v16h A1 = load_af32(rows, 1, lane);

#pragma unroll
    for (int nt = 0; nt < 4; ++nt) {
        v8f cs = zero8();
        v8f cr = zero8();
        cs = wmma16(A0, load_bp(Pm1s, nt, 0, lane), cs);
        cs = wmma16(A1, load_bp(Pm1s, nt, 1, lane), cs);
        cr = wmma16(A0, load_bp(Pm1r, nt, 0, lane), cr);
        cr = wmma16(A1, load_bp(Pm1r, nt, 1, lane), cr);
#pragma unroll
        for (int r = 0; r < 8; ++r) {
            int m = r + 8 * hs;
            int idx = (tile * 16 + m) * DD + nt * 16 + n;
            Hs[idx] = cs[r];
            Hr[idx] = cr[r];
        }
    }
}

// ---------------------------------------------------------------------------
// agg[b,i,:] = (1/N) * sum_j adj[b,i,j] * tanh( tanh(Hs[j]+Hr[i]+b1) @ Wm2 + b2 )
// grid 1536 (one wave per (b,i))
__global__ void k_pair(const float* __restrict__ Hs, const float* __restrict__ Hr,
                       const float* __restrict__ b1, const _Float16* __restrict__ Pm2,
                       const float* __restrict__ b2, const float* __restrict__ adj,
                       float* __restrict__ agg) {
    int pair = blockIdx.x;
    int b = pair / NN, i = pair % NN;
    int lane = threadIdx.x;
    int n = lane & 15, hs = lane >> 4;

    __shared__ float hrb[DD];
    __shared__ float sb2[DD];
    __shared__ _Float16 atile[16 * DD];

    for (int k = lane; k < DD; k += 32) {
        hrb[k] = Hr[(b * NN + i) * DD + k] + b1[k];
        sb2[k] = b2[k];
    }
    __syncthreads();

    v16h B[4][2];
#pragma unroll
    for (int nt = 0; nt < 4; ++nt)
#pragma unroll
        for (int kh = 0; kh < 2; ++kh)
            B[nt][kh] = load_bp(Pm2, nt, kh, lane);

    float acc[4] = {0.f, 0.f, 0.f, 0.f};
    const float* adjrow = adj + (b * NN + i) * NN;

    int rr = lane >> 1;     // staging row 0..15
    int hh = lane & 1;      // staging column half

    for (int jt = 0; jt < 6; ++jt) {
        // stage A = tanh(Hs[j] + Hr[i] + b1) -> LDS f16 row-major, coalesced b128
        const float* hsrow = Hs + (b * NN + jt * 16 + rr) * DD + hh * 32;
        const float* hb = hrb + hh * 32;
        _Float16* orow = atile + rr * DD + hh * 32;
#pragma unroll
        for (int q = 0; q < 4; ++q) {
            f4 x0 = *(const f4*)(hsrow + q * 8);
            f4 x1 = *(const f4*)(hsrow + q * 8 + 4);
            h8 o;
#pragma unroll
            for (int e = 0; e < 4; ++e) {
                o[e]     = (_Float16)tanhf(x0[e] + hb[q * 8 + e]);
                o[e + 4] = (_Float16)tanhf(x1[e] + hb[q * 8 + 4 + e]);
            }
            *(h8*)(orow + q * 8) = o;
        }
        __syncthreads();

        v16h A0 = load_alds(atile, 0, lane);
        v16h A1 = load_alds(atile, 1, lane);

        f4 aw0 = *(const f4*)(adjrow + jt * 16 + 8 * hs);
        f4 aw1 = *(const f4*)(adjrow + jt * 16 + 8 * hs + 4);

#pragma unroll
        for (int nt = 0; nt < 4; ++nt) {
            v8f c = zero8();
            c = wmma16(A0, B[nt][0], c);
            c = wmma16(A1, B[nt][1], c);
            float bb = sb2[nt * 16 + n];
#pragma unroll
            for (int r = 0; r < 4; ++r) {
                acc[nt] += aw0[r] * tanhf(c[r] + bb);
                acc[nt] += aw1[r] * tanhf(c[r + 4] + bb);
            }
        }
        __syncthreads();
    }

#pragma unroll
    for (int nt = 0; nt < 4; ++nt) {
        float v = acc[nt] + __shfl_xor(acc[nt], 16, 32);
        if (lane < 16)
            agg[(b * NN + i) * DD + nt * 16 + lane] = v * (1.0f / (float)NN);
    }
}

// ---------------------------------------------------------------------------
// xr/xi/xn[b,d] = emb[skill_t[b]] @ W_i* + b_i*
__global__ void k_x(const int* __restrict__ skill, const float* __restrict__ emb,
                    const float* __restrict__ Wir, const float* __restrict__ bir,
                    const float* __restrict__ Wii, const float* __restrict__ bii,
                    const float* __restrict__ Win, const float* __restrict__ bin_,
                    float* __restrict__ xr, float* __restrict__ xi, float* __restrict__ xn,
                    int t) {
    int b = blockIdx.x;
    int d = threadIdx.x;
    int s = skill[b * TT + t];
    const float* e = emb + s * DD;
    float sr = bir[d], si = bii[d], sn = bin_[d];
    for (int k = 0; k < DD; ++k) {
        float v = e[k];
        sr += v * Wir[k * DD + d];
        si += v * Wii[k * DD + d];
        sn += v * Win[k * DD + d];
    }
    xr[b * DD + d] = sr;
    xi[b * DD + d] = si;
    xn[b * DD + d] = sn;
}

// ---------------------------------------------------------------------------
// GRU update from agg@{W_hr,W_hi,W_hh} + x*, hidden updated in place
__global__ void k_gru(const float* __restrict__ agg,
                      const _Float16* __restrict__ Phr, const _Float16* __restrict__ Phi,
                      const _Float16* __restrict__ Phh,
                      const float* __restrict__ xr, const float* __restrict__ xi,
                      const float* __restrict__ xn,
                      float* __restrict__ hidden) {
    int tile = blockIdx.x;
    int lane = threadIdx.x;
    int n = lane & 15, hs = lane >> 4;
    int b = tile / 6;

    const float* rows = agg + tile * 16 * DD;
    v16h A0 = load_af32(rows, 0, lane);
    v16h A1 = load_af32(rows, 1, lane);

    v8f Cr[4], Ci[4], Ch[4];
#pragma unroll
    for (int nt = 0; nt < 4; ++nt) {
        Cr[nt] = zero8(); Ci[nt] = zero8(); Ch[nt] = zero8();
        Cr[nt] = wmma16(A0, load_bp(Phr, nt, 0, lane), Cr[nt]);
        Cr[nt] = wmma16(A1, load_bp(Phr, nt, 1, lane), Cr[nt]);
        Ci[nt] = wmma16(A0, load_bp(Phi, nt, 0, lane), Ci[nt]);
        Ci[nt] = wmma16(A1, load_bp(Phi, nt, 1, lane), Ci[nt]);
        Ch[nt] = wmma16(A0, load_bp(Phh, nt, 0, lane), Ch[nt]);
        Ch[nt] = wmma16(A1, load_bp(Phh, nt, 1, lane), Ch[nt]);
    }

#pragma unroll
    for (int nt = 0; nt < 4; ++nt) {
#pragma unroll
        for (int r = 0; r < 8; ++r) {
            int m = r + 8 * hs;
            int row = tile * 16 + m;
            int d = nt * 16 + n;
            float rg = 1.0f / (1.0f + expf(-(xr[b * DD + d] + Cr[nt][r])));
            float ig = 1.0f / (1.0f + expf(-(xi[b * DD + d] + Ci[nt][r])));
            float ng = tanhf(xn[b * DD + d] + rg * Ch[nt][r]);
            float h  = hidden[row * DD + d];
            hidden[row * DD + d] = (1.0f - ig) * ng + ig * h;
        }
    }
}

// ---------------------------------------------------------------------------
// Output MLP, staged through f16 LDS between layers
__global__ void k_mlp(const float* __restrict__ hidden,
                      const _Float16* __restrict__ Po1, const float* __restrict__ bo1,
                      const _Float16* __restrict__ Po2, const float* __restrict__ bo2,
                      const _Float16* __restrict__ Po3, const float* __restrict__ bo3,
                      float* __restrict__ out, int t) {
    int tile = blockIdx.x;
    int lane = threadIdx.x;
    int n = lane & 15, hs = lane >> 4;
    __shared__ _Float16 buf[16 * DD];

    v16h A0 = load_af32(hidden + tile * 16 * DD, 0, lane);
    v16h A1 = load_af32(hidden + tile * 16 * DD, 1, lane);

    // layer 1 -> buf (f16)
#pragma unroll
    for (int nt = 0; nt < 4; ++nt) {
        v8f c = zero8();
        c = wmma16(A0, load_bp(Po1, nt, 0, lane), c);
        c = wmma16(A1, load_bp(Po1, nt, 1, lane), c);
        float bb = bo1[nt * 16 + n];
#pragma unroll
        for (int r = 0; r < 8; ++r) {
            int m = r + 8 * hs;
            buf[m * DD + nt * 16 + n] = (_Float16)fmaxf(c[r] + bb, 0.0f);
        }
    }
    __syncthreads();
    A0 = load_alds(buf, 0, lane);
    A1 = load_alds(buf, 1, lane);
    __syncthreads();

    // layer 2 -> buf (f16)
#pragma unroll
    for (int nt = 0; nt < 4; ++nt) {
        v8f c = zero8();
        c = wmma16(A0, load_bp(Po2, nt, 0, lane), c);
        c = wmma16(A1, load_bp(Po2, nt, 1, lane), c);
        float bb = bo2[nt * 16 + n];
#pragma unroll
        for (int r = 0; r < 8; ++r) {
            int m = r + 8 * hs;
            buf[m * DD + nt * 16 + n] = (_Float16)fmaxf(c[r] + bb, 0.0f);
        }
    }
    __syncthreads();
    A0 = load_alds(buf, 0, lane);
    A1 = load_alds(buf, 1, lane);

    // layer 3 -> out (f32)
#pragma unroll
    for (int nt = 0; nt < 4; ++nt) {
        v8f c = zero8();
        c = wmma16(A0, load_bp(Po3, nt, 0, lane), c);
        c = wmma16(A1, load_bp(Po3, nt, 1, lane), c);
        float bb = bo3[nt * 16 + n];
#pragma unroll
        for (int r = 0; r < 8; ++r) {
            int m = r + 8 * hs;
            int row = tile * 16 + m;     // = b*96 + node
            int bi = row / NN;
            int node = row % NN;
            int d = nt * 16 + n;
            out[(((size_t)bi * (TT - 1) + t) * NN + node) * DD + d] = c[r] + bb;
        }
    }
}

// ---------------------------------------------------------------------------
extern "C" void kernel_launch(void* const* d_in, const int* in_sizes, int n_in,
                              void* d_out, int out_size, void* d_ws, size_t ws_size,
                              hipStream_t stream) {
    (void)in_sizes; (void)n_in; (void)out_size; (void)ws_size;

    const int*   skill = (const int*)d_in[0];
    const float* adj   = (const float*)d_in[1];
    const float* emb   = (const float*)d_in[2];
    const float* Wm1   = (const float*)d_in[3];
    const float* b1    = (const float*)d_in[4];
    const float* Wm2   = (const float*)d_in[5];
    const float* b2    = (const float*)d_in[6];
    const float* Whr   = (const float*)d_in[7];
    const float* Whi   = (const float*)d_in[8];
    const float* Whh   = (const float*)d_in[9];
    const float* Wir   = (const float*)d_in[10];
    const float* bir   = (const float*)d_in[11];
    const float* Wii   = (const float*)d_in[12];
    const float* bii   = (const float*)d_in[13];
    const float* Win   = (const float*)d_in[14];
    const float* bin_  = (const float*)d_in[15];
    const float* Wo1   = (const float*)d_in[16];
    const float* bo1   = (const float*)d_in[17];
    const float* Wo2   = (const float*)d_in[18];
    const float* bo2   = (const float*)d_in[19];
    const float* Wo3   = (const float*)d_in[20];
    const float* bo3   = (const float*)d_in[21];
    float* out = (float*)d_out;

    float* ws     = (float*)d_ws;
    float* hidden = ws;                        // 1536*64
    float* Hs     = hidden + ROWS * DD;
    float* Hr     = Hs     + ROWS * DD;
    float* agg    = Hr     + ROWS * DD;
    float* xr     = agg    + ROWS * DD;        // 16*64 x3
    float* xi     = xr     + BSZ * DD;
    float* xn     = xi     + BSZ * DD;
    _Float16* packW = (_Float16*)(xn + BSZ * DD);   // 9 * 4096 halves = 72KB

    const _Float16* Pm1s = packW + 0 * WSTR;
    const _Float16* Pm1r = packW + 1 * WSTR;
    const _Float16* Pm2  = packW + 2 * WSTR;
    const _Float16* Phr  = packW + 3 * WSTR;
    const _Float16* Phi  = packW + 4 * WSTR;
    const _Float16* Phh  = packW + 5 * WSTR;
    const _Float16* Po1  = packW + 6 * WSTR;
    const _Float16* Po2  = packW + 7 * WSTR;
    const _Float16* Po3  = packW + 8 * WSTR;

    hipMemsetAsync(hidden, 0, (size_t)ROWS * DD * sizeof(float), stream);
    k_pack<<<72, 32, 0, stream>>>(Wm1, Wm2, Whr, Whi, Whh, Wo1, Wo2, Wo3, packW);

    for (int t = 0; t < TT - 1; ++t) {
        k_hsr<<<ROWS / 16, 32, 0, stream>>>(hidden, Pm1s, Pm1r, Hs, Hr);
        k_pair<<<ROWS, 32, 0, stream>>>(Hs, Hr, b1, Pm2, b2, adj, agg);
        k_x<<<BSZ, DD, 0, stream>>>(skill, emb, Wir, bir, Wii, bii, Win, bin_,
                                    xr, xi, xn, t);
        k_gru<<<ROWS / 16, 32, 0, stream>>>(agg, Phr, Phi, Phh, xr, xi, xn, hidden);
        k_mlp<<<ROWS / 16, 32, 0, stream>>>(hidden, Po1, bo1, Po2, bo2, Po3, bo3,
                                            out, t);
    }
}
